// LoraLinear_22600117911582
// MI455X (gfx1250) — matrix operand order repaired and checked by reading, not compiled, not run on hardware
//
#include <hip/hip_runtime.h>
#include <string.h>

typedef __attribute__((ext_vector_type(16))) _Float16 v16h;
typedef __attribute__((ext_vector_type(8)))  float    v8f;
typedef int vi4 __attribute__((vector_size(16)));

#define M_DIM 8192
#define N_DIM 4096
#define K_DIM 4096
#define R_DIM 64

#define BK 64
#define LDS_STRIDE 80   // 64 halves + pad -> 160B row stride, 16B aligned

#define AS1 __attribute__((address_space(1)))
#define AS3 __attribute__((address_space(3)))

#if __has_builtin(__builtin_amdgcn_global_load_async_to_lds_b128) && \
    __has_builtin(__builtin_amdgcn_s_wait_asynccnt)
#define USE_ASYNC_LDS 1
#else
#define USE_ASYNC_LDS 0
#endif

union Frag {
    v16h h;
    uint4 q[2];
};

__device__ __forceinline__ v8f vzero8() {
    v8f z = {0.f,0.f,0.f,0.f,0.f,0.f,0.f,0.f};
    return z;
}

// ---------------------------------------------------------------------------
// f32 -> f16 converter (4 elements/thread)
// ---------------------------------------------------------------------------
struct h4 { _Float16 a, b, c, d; };

__global__ void cvt_f32_to_f16(const float* __restrict__ src,
                               _Float16* __restrict__ dst, int n4) {
    int i = blockIdx.x * blockDim.x + threadIdx.x;
    if (i < n4) {
        float4 v = reinterpret_cast<const float4*>(src)[i];
        h4 o;
        o.a = (_Float16)v.x; o.b = (_Float16)v.y;
        o.c = (_Float16)v.z; o.d = (_Float16)v.w;
        reinterpret_cast<h4*>(dst)[i] = o;
    }
}

// int32 codes -> f16 (exact for |code| <= 127)
__global__ void cvt_i32_to_f16(const int* __restrict__ src,
                               _Float16* __restrict__ dst, int n4) {
    int i = blockIdx.x * blockDim.x + threadIdx.x;
    if (i < n4) {
        int4 v = reinterpret_cast<const int4*>(src)[i];
        h4 o;
        o.a = (_Float16)(float)v.x; o.b = (_Float16)(float)v.y;
        o.c = (_Float16)(float)v.z; o.d = (_Float16)(float)v.w;
        reinterpret_cast<h4*>(dst)[i] = o;
    }
}

// ---------------------------------------------------------------------------
// xa_h[m, r] = f16( 2 * sum_k Xh[m,k] * A[r,k] )   (one block per m row)
// ---------------------------------------------------------------------------
__global__ void lora_xa_kernel(const _Float16* __restrict__ Xh,
                               const float* __restrict__ loraA,
                               _Float16* __restrict__ xah) {
    __shared__ float xbuf[512];
    __shared__ float red[4][64];

    const int m   = blockIdx.x;
    const int tid = threadIdx.x;
    const int r   = tid & 63;
    const int seg = tid >> 6;

    const _Float16* xrow = Xh + (size_t)m * K_DIM;
    const float*    arow = loraA + (size_t)r * K_DIM;

    float acc = 0.f;
    for (int k0 = 0; k0 < K_DIM; k0 += 512) {
        __syncthreads();
        unsigned u = *reinterpret_cast<const unsigned*>(xrow + k0 + 2 * tid);
        _Float16 h0, h1;
        memcpy(&h0, &u, 2);
        unsigned uh = u >> 16;
        memcpy(&h1, &uh, 2);
        xbuf[2 * tid]     = (float)h0;
        xbuf[2 * tid + 1] = (float)h1;
        __syncthreads();
        const int kb = seg * 128;
        #pragma unroll 8
        for (int k = 0; k < 128; ++k)
            acc += xbuf[kb + k] * arow[k0 + kb + k];
    }
    __syncthreads();
    red[seg][r] = acc;
    __syncthreads();
    if (tid < 64) {
        float s = red[0][tid] + red[1][tid] + red[2][tid] + red[3][tid];
        xah[(size_t)m * R_DIM + tid] = (_Float16)(2.0f * s);  // fold SCALING
    }
}

// ---------------------------------------------------------------------------
// Main fused GEMM, double-buffered LDS, async global->LDS when available:
//   out[m,n] = scale[n] * (Xh @ Wh^T)[m,n] + (xah @ Bh^T)[m,n]
// Block tile 128x128, BK=64, 8 waves (wave32), wave tile 64x32 (4x2 WMMA).
// ---------------------------------------------------------------------------
__global__ __launch_bounds__(256)
void lora_gemm_kernel(const _Float16* __restrict__ Xh,
                      const _Float16* __restrict__ Wh,
                      const float* __restrict__ scale,
                      const _Float16* __restrict__ xah,
                      const _Float16* __restrict__ Bh,
                      float* __restrict__ out) {
    __shared__ _Float16 lA[2][128][LDS_STRIDE];
    __shared__ _Float16 lB[2][128][LDS_STRIDE];

    const int tid   = threadIdx.x;
    const int wave  = tid >> 5;
    const int lane  = tid & 31;
    const int lrow  = lane & 15;
    const int lhalf = lane >> 4;

    const int wm = wave & 1;            // 2 waves along M
    const int wn = wave >> 1;           // 4 waves along N
    const int aoff = wm * 64;
    const int noff = wn * 32;

    const int bx = blockIdx.x;          // N tile
    const int by = blockIdx.y;          // M tile

    const int grow   = tid >> 3;        // 32 rows per pass
    const int gchunk = tid & 7;         // 8 x 16B per row

    const _Float16* asrc = Xh + (size_t)(by * 128) * K_DIM;
    const _Float16* bsrc = Wh + (size_t)(bx * 128) * K_DIM;

    v8f acc[4][2];
    #pragma unroll
    for (int mi = 0; mi < 4; ++mi)
        #pragma unroll
        for (int ni = 0; ni < 2; ++ni)
            acc[mi][ni] = vzero8();

#if USE_ASYNC_LDS
    auto stage_async = [&](int buf, int k0) {
        #pragma unroll
        for (int p = 0; p < 4; ++p) {
            const int r = grow + 32 * p;
            const _Float16* ga = asrc + (size_t)r * K_DIM + k0 + gchunk * 8;
            const _Float16* gb = bsrc + (size_t)r * K_DIM + k0 + gchunk * 8;
            _Float16* la = &lA[buf][r][gchunk * 8];
            _Float16* lb = &lB[buf][r][gchunk * 8];
            __builtin_amdgcn_global_load_async_to_lds_b128(
                (AS1 vi4*)(uintptr_t)ga,
                (AS3 vi4*)(unsigned)(uintptr_t)la, 0, 0);
            __builtin_amdgcn_global_load_async_to_lds_b128(
                (AS1 vi4*)(uintptr_t)gb,
                (AS3 vi4*)(unsigned)(uintptr_t)lb, 0, 0);
        }
    };
#else
    uint4 ra[4], rb[4];
    auto load_regs = [&](int k0) {
        #pragma unroll
        for (int p = 0; p < 4; ++p) {
            const int r = grow + 32 * p;
            ra[p] = *reinterpret_cast<const uint4*>(asrc + (size_t)r * K_DIM + k0 + gchunk * 8);
            rb[p] = *reinterpret_cast<const uint4*>(bsrc + (size_t)r * K_DIM + k0 + gchunk * 8);
        }
    };
    auto store_regs = [&](int buf) {
        #pragma unroll
        for (int p = 0; p < 4; ++p) {
            const int r = grow + 32 * p;
            *reinterpret_cast<uint4*>(&lA[buf][r][gchunk * 8]) = ra[p];
            *reinterpret_cast<uint4*>(&lB[buf][r][gchunk * 8]) = rb[p];
        }
    };
#endif

    auto compute = [&](int buf) {
        #pragma unroll
        for (int kk = 0; kk < BK; kk += 32) {
            Frag a[4], b[2];
            #pragma unroll
            for (int mi = 0; mi < 4; ++mi) {
                const _Float16* p = &lA[buf][aoff + mi * 16 + lrow][kk + 8 * lhalf];
                a[mi].q[0] = *reinterpret_cast<const uint4*>(p);
                a[mi].q[1] = *reinterpret_cast<const uint4*>(p + 16);
            }
            #pragma unroll
            for (int ni = 0; ni < 2; ++ni) {
                const _Float16* p = &lB[buf][noff + ni * 16 + lrow][kk + 8 * lhalf];
                b[ni].q[0] = *reinterpret_cast<const uint4*>(p);
                b[ni].q[1] = *reinterpret_cast<const uint4*>(p + 16);
            }
            #pragma unroll
            for (int mi = 0; mi < 4; ++mi)
                #pragma unroll
                for (int ni = 0; ni < 2; ++ni)
                    acc[mi][ni] = __builtin_amdgcn_wmma_f32_16x16x32_f16(
                        false, a[mi].h, false, b[ni].h,
                        (short)0, acc[mi][ni], false, false);
        }
    };

    // ---- software-pipelined main loop over K ----
#if USE_ASYNC_LDS
    stage_async(0, 0);
    __builtin_amdgcn_s_wait_asynccnt(0);
    __syncthreads();
    for (int s = 1; s < K_DIM / BK; ++s) {
        const int k0 = s * BK;
        stage_async(s & 1, k0);
        if (k0 + BK < K_DIM) {
            __builtin_prefetch(asrc + (size_t)grow * K_DIM + k0 + BK, 0, 1);
            __builtin_prefetch(bsrc + (size_t)grow * K_DIM + k0 + BK, 0, 1);
        }
        compute((s - 1) & 1);
        __builtin_amdgcn_s_wait_asynccnt(0);
        __syncthreads();
    }
    compute(((K_DIM / BK) - 1) & 1);
#else
    load_regs(0);
    store_regs(0);
    __syncthreads();
    for (int s = 1; s < K_DIM / BK; ++s) {
        const int k0 = s * BK;
        load_regs(k0);
        if (k0 + BK < K_DIM) {
            __builtin_prefetch(asrc + (size_t)grow * K_DIM + k0 + BK, 0, 1);
            __builtin_prefetch(bsrc + (size_t)grow * K_DIM + k0 + BK, 0, 1);
        }
        compute((s - 1) & 1);
        store_regs(s & 1);
        __syncthreads();
    }
    compute(((K_DIM / BK) - 1) & 1);
#endif

    // ---- epilogue: per-column dequant scale on the base GEMM ----
    int ncol[2];
    #pragma unroll
    for (int ni = 0; ni < 2; ++ni) {
        ncol[ni] = bx * 128 + noff + ni * 16 + lrow;
        const float s = scale[ncol[ni]];
        #pragma unroll
        for (int mi = 0; mi < 4; ++mi)
            #pragma unroll
            for (int e = 0; e < 8; ++e)
                acc[mi][ni][e] *= s;
    }

    // ---- fold LoRA rank-64 update: two more WMMA k-steps from global ----
    #pragma unroll
    for (int kk = 0; kk < R_DIM; kk += 32) {
        Frag a[4], b[2];
        #pragma unroll
        for (int mi = 0; mi < 4; ++mi) {
            const int mrow = by * 128 + aoff + mi * 16 + lrow;
            const _Float16* p = xah + (size_t)mrow * R_DIM + kk + 8 * lhalf;
            a[mi].q[0] = *reinterpret_cast<const uint4*>(p);
            a[mi].q[1] = *reinterpret_cast<const uint4*>(p + 16);
        }
        #pragma unroll
        for (int ni = 0; ni < 2; ++ni) {
            const _Float16* p = Bh + (size_t)ncol[ni] * R_DIM + kk + 8 * lhalf;
            b[ni].q[0] = *reinterpret_cast<const uint4*>(p);
            b[ni].q[1] = *reinterpret_cast<const uint4*>(p + 16);
        }
        #pragma unroll
        for (int mi = 0; mi < 4; ++mi)
            #pragma unroll
            for (int ni = 0; ni < 2; ++ni)
                acc[mi][ni] = __builtin_amdgcn_wmma_f32_16x16x32_f16(
                    false, a[mi].h, false, b[ni].h,
                    (short)0, acc[mi][ni], false, false);
    }

    // ---- store ----
    #pragma unroll
    for (int mi = 0; mi < 4; ++mi) {
        const int mbase = by * 128 + aoff + mi * 16 + 8 * lhalf;
        #pragma unroll
        for (int ni = 0; ni < 2; ++ni) {
            #pragma unroll
            for (int r = 0; r < 8; ++r)
                out[(size_t)(mbase + r) * N_DIM + ncol[ni]] = acc[mi][ni][r];
        }
    }
}

// ---------------------------------------------------------------------------
extern "C" void kernel_launch(void* const* d_in, const int* in_sizes, int n_in,
                              void* d_out, int out_size, void* d_ws, size_t ws_size,
                              hipStream_t stream) {
    const float* x      = (const float*)d_in[0];
    const int*   qw     = (const int*)d_in[1];
    const float* scale  = (const float*)d_in[2];
    const float* loraA  = (const float*)d_in[3];
    const float* loraB  = (const float*)d_in[4];
    float* out          = (float*)d_out;

    char* ws = (char*)d_ws;
    _Float16* Xh  = (_Float16*)(ws);                        // 64 MB
    _Float16* Wh  = (_Float16*)(ws + 67108864);             // 32 MB
    _Float16* Bh  = (_Float16*)(ws + 100663296);            // 0.5 MB
    _Float16* xah = (_Float16*)(ws + 101187584);            // 1 MB

    {
        int n4 = (M_DIM * K_DIM) / 4;
        cvt_f32_to_f16<<<n4 / 256, 256, 0, stream>>>(x, Xh, n4);
    }
    {
        int n4 = (N_DIM * K_DIM) / 4;
        cvt_i32_to_f16<<<n4 / 256, 256, 0, stream>>>(qw, Wh, n4);
    }
    {
        int n4 = (N_DIM * R_DIM) / 4;
        cvt_f32_to_f16<<<n4 / 256, 256, 0, stream>>>(loraB, Bh, n4);
    }
    lora_xa_kernel<<<M_DIM, 256, 0, stream>>>(Xh, loraA, xah);

    dim3 grid(N_DIM / 128, M_DIM / 128);
    lora_gemm_kernel<<<grid, 256, 0, stream>>>(Xh, Wh, scale, xah, Bh, out);
}
